// FeatureFusionModule_85813446574859
// MI455X (gfx1250) — matrix operand order repaired and checked by reading, not compiled, not run on hardware
//
#include <hip/hip_runtime.h>
#include <hip/hip_bf16.h>

// CDNA5 / gfx1250. wave32. v_wmma_f32_16x16x32_f16 + double-buffered
// async-to-LDS staging (ASYNCcnt) overlapping copy with WMMA compute.

typedef __attribute__((ext_vector_type(16))) _Float16 v16h;
typedef __attribute__((ext_vector_type(8)))  float    v8f;
typedef __attribute__((ext_vector_type(4)))  int      v4i;

union Frag {
    uint4        q[2];
    unsigned int u[8];
    v16h         v;
};

__device__ __forceinline__ unsigned int pkh(float a, float b) {
    union { _Float16 h[2]; unsigned int u; } p;
    p.h[0] = (_Float16)a;
    p.h[1] = (_Float16)b;
    return p.u;
}

__device__ __forceinline__ void ldB(Frag& B, const unsigned int* bp) {
    B.q[0] = *(const uint4*)bp;
    B.q[1] = *(const uint4*)(bp + 8);
}

// ---- CDNA5 async global->LDS copy (ASYNCcnt path), with safe fallback ------
#if defined(__gfx1250__) && __has_builtin(__builtin_amdgcn_global_load_async_to_lds_b128)
#define FFM_ASYNC 1
#else
#define FFM_ASYNC 0
#endif

__device__ __forceinline__ void cp16(const unsigned int* g, unsigned int* l) {
#if FFM_ASYNC
    __builtin_amdgcn_global_load_async_to_lds_b128(
        (__attribute__((address_space(1))) v4i*)(g),
        (__attribute__((address_space(3))) v4i*)(l), 0, 0);
#else
    *(uint4*)l = *(const uint4*)g;
#endif
}

__device__ __forceinline__ void async_wait0() {
#if FFM_ASYNC
#if __has_builtin(__builtin_amdgcn_s_wait_asynccnt)
    __builtin_amdgcn_s_wait_asynccnt(0);
#else
    asm volatile("s_wait_asynccnt 0" ::: "memory");
#endif
#endif
}

// ---------------------------------------------------------------------------
// Weight repack: conv_w [128][1024][3][3] f32 ->
//   wp2[((kc*128 + o)*9 + tap)*16 + q] = pack_f16(w[o][kc*32+2q][tap], w[o][kc*32+2q+1][tap])
// ---------------------------------------------------------------------------
__global__ void ffm_repack2(const float* __restrict__ w2, unsigned int* __restrict__ wp2) {
    int idx = blockIdx.x * 256 + threadIdx.x;
    if (idx >= 32 * 128 * 9 * 16) return;
    int q    = idx & 15;
    int rest = idx >> 4;
    int tap  = rest % 9;   rest /= 9;
    int o    = rest & 127;
    int kc   = rest >> 7;
    int c    = kc * 32 + 2 * q;
    float f0 = w2[(long)(o * 1024 + c)     * 9 + tap];
    float f1 = w2[(long)(o * 1024 + c + 1) * 9 + tap];
    wp2[idx] = pkh(f0, f1);
}

// conv1_w [128][128][7] -> wp1[((kc*128 + o)*7 + tap)*16 + q]
__global__ void ffm_repack1(const float* __restrict__ w1, unsigned int* __restrict__ wp1) {
    int idx = blockIdx.x * 256 + threadIdx.x;
    if (idx >= 4 * 128 * 7 * 16) return;
    int q    = idx & 15;
    int rest = idx >> 4;
    int tap  = rest % 7;   rest /= 7;
    int o    = rest & 127;
    int kc   = rest >> 7;
    int c    = kc * 32 + 2 * q;
    float f0 = w1[(long)(o * 128 + c)     * 7 + tap];
    float f1 = w1[(long)(o * 128 + c + 1) * 7 + tap];
    wp1[idx] = pkh(f0, f1);
}

// ---------------------------------------------------------------------------
// Input pre-pack (one shot): concat(in1,in2) f32 NCHW ->
//   xp[(((b*32+kc)*64 + row)*128 + w)*16 + q] = pack_f16(x[c=kc*32+2q], x[c+1])
// ---------------------------------------------------------------------------
__global__ __launch_bounds__(256) void ffm_packin(
    const float* __restrict__ in1, const float* __restrict__ in2,
    unsigned int* __restrict__ xp)
{
    __shared__ _Float16 t[32 * 128];
    const int blk = blockIdx.x;        // ((b*32 + kc)*64 + row)
    const int row = blk & 63;
    const int rst = blk >> 6;
    const int kc  = rst & 31;
    const int b   = rst >> 5;
    const int tid = threadIdx.x;

    for (int idx = tid; idx < 32 * 128; idx += 256) {
        int cc = idx >> 7;
        int w  = idx & 127;
        int c  = kc * 32 + cc;
        const float* src = (c < 512)
            ? in1 + (((long)(b * 512 + c)         * 64 + row) * 128 + w)
            : in2 + (((long)(b * 512 + (c - 512)) * 64 + row) * 128 + w);
        t[idx] = (_Float16)(*src);
    }
    __syncthreads();
    unsigned int* dst = xp + (long)blk * 2048;
    for (int idx = tid; idx < 128 * 16; idx += 256) {
        int w = idx >> 4, q = idx & 15;
        union { _Float16 h[2]; unsigned int u; } p;
        p.h[0] = t[(2 * q)     * 128 + w];
        p.h[1] = t[(2 * q + 1) * 128 + w];
        dst[idx] = p.u;
    }
}

// issue async copies for one 32-channel chunk into one LDS buffer
__device__ __forceinline__ void ffm_stage(
    const unsigned int* __restrict__ xp, unsigned int* dst,
    int b, int kcc, int h, int tid)
{
    const unsigned int* src = xp + (long)(b * 32 + kcc) * 64 * 2048;
    for (int idx = tid; idx < 3 * 128 * 4; idx += 256) {   // 6 iters of 16B quads
        int r   = idx / 512;
        int rem = idx - r * 512;
        int w   = rem >> 2, qq = rem & 3;
        int row = h - 1 + r;
        if ((unsigned)row < 64u)
            cp16(src + ((long)row * 2048 + w * 16 + qq * 4),
                 &dst[(r * 130 + w + 1) * 16 + qq * 4]);
    }
}

// ---------------------------------------------------------------------------
// Conv2d (packed-input): 3x3, 1024->128, pad 1, + BN + ReLU.
// Block = (h, b); 8 waves; wave = (32-channel M-group) x (64-column N-half).
// Double-buffered LDS: async-copy chunk kc+1 while WMMAs consume chunk kc.
// Each B fragment feeds TWO WMMAs (A0, A1).
// ---------------------------------------------------------------------------
#define LBUF (3 * 130 * 16)            // 6240 uints per buffer (16B-aligned size)

__global__ __launch_bounds__(256) void ffm_conv2d_pk(
    const unsigned int* __restrict__ xp,
    const unsigned int* __restrict__ wp2,
    const float* __restrict__ cb,  const float* __restrict__ g,
    const float* __restrict__ be,  const float* __restrict__ mu,
    const float* __restrict__ var, float* __restrict__ out)
{
    __shared__ alignas(16) unsigned int ldsi[2 * LBUF];
    const int b    = blockIdx.y;
    const int h    = blockIdx.x;
    const int tid  = threadIdx.x;
    const int lane = tid & 31;
    const int wv   = tid >> 5;
    const int mg   = wv & 3;
    const int nh   = wv >> 2;
    const int o0   = mg * 32;
    const int nc0  = nh * 4;           // first N-tile index
    const int l15  = lane & 15;
    const int hi   = lane >> 4;
    const int qa   = hi ? 4 : 0;
    const int qb   = hi ? 4 : 0;

    // zero padding regions once in BOTH buffers (never overwritten by copies)
    for (int idx = tid; idx < LBUF; idx += 256) {
        int p   = idx >> 4;
        int r   = p / 130;
        int col = p - r * 130;
        int row = h - 1 + r;
        if (!((unsigned)row < 64u) || col == 0 || col == 129) {
            ldsi[idx]        = 0u;
            ldsi[idx + LBUF] = 0u;
        }
    }

    v8f acc0[4], acc1[4];
#pragma unroll
    for (int t = 0; t < 4; ++t) { acc0[t] = {}; acc1[t] = {}; }

    // prologue: stage chunk 0
    ffm_stage(xp, ldsi, b, 0, h, tid);
    async_wait0();
    __syncthreads();

    for (int kc = 0; kc < 32; ++kc) {
        unsigned int* cur = ldsi + (kc & 1) * LBUF;
        if (kc + 1 < 32)               // overlap: copy kc+1 under kc's WMMAs
            ffm_stage(xp, ldsi + ((kc + 1) & 1) * LBUF, b, kc + 1, h, tid);

#pragma unroll
        for (int tap = 0; tap < 9; ++tap) {
            const int dh = tap / 3, dw = tap % 3;
            Frag A0, A1;
            const unsigned int* wp = wp2 + (long)(kc * 128 + o0 + l15) * 144 + tap * 16;
            A0.q[0] = *(const uint4*)(wp + qa);
            A0.q[1] = *(const uint4*)(wp + qa + 8);
            A1.q[0] = *(const uint4*)(wp + 16 * 144 + qa);
            A1.q[1] = *(const uint4*)(wp + 16 * 144 + qa + 8);
            const int rb = dh * 130;
#pragma unroll
            for (int nt = 0; nt < 4; ++nt) {
                Frag B;
                ldB(B, &cur[(rb + (nc0 + nt) * 16 + l15 + dw) * 16 + qb]);
                acc0[nt] = __builtin_amdgcn_wmma_f32_16x16x32_f16(
                    false, A0.v, false, B.v, (short)0, acc0[nt], false, false);
                acc1[nt] = __builtin_amdgcn_wmma_f32_16x16x32_f16(
                    false, A1.v, false, B.v, (short)0, acc1[nt], false, false);
            }
        }

        async_wait0();                 // next buffer's copies landed
        __syncthreads();
    }

#pragma unroll
    for (int a = 0; a < 2; ++a) {
#pragma unroll
        for (int r = 0; r < 8; ++r) {
            const int o   = o0 + a * 16 + r + hi * 8;
            const float inv = g[o] * rsqrtf(var[o] + 1e-5f);
            const float sh  = (cb[o] - mu[o]) * inv + be[o];
            float* orow = out + ((long)(b * 128 + o) * 64 + h) * 128;
#pragma unroll
            for (int nt = 0; nt < 4; ++nt) {
                float v = (a ? acc1[nt][r] : acc0[nt][r]) * inv + sh;
                orow[(nc0 + nt) * 16 + l15] = v > 0.f ? v : 0.f;
            }
        }
    }
}

// ---------------------------------------------------------------------------
// Conv2d fallback (stages f32 + converts in-kernel; used if ws can't hold xp)
// ---------------------------------------------------------------------------
__global__ __launch_bounds__(256) void ffm_conv2d(
    const float* __restrict__ in1, const float* __restrict__ in2,
    const unsigned int* __restrict__ wp2,
    const float* __restrict__ cb,  const float* __restrict__ g,
    const float* __restrict__ be,  const float* __restrict__ mu,
    const float* __restrict__ var, float* __restrict__ out)
{
    __shared__ alignas(16) unsigned int ldsi[3 * 130 * 16];
    const int b    = blockIdx.y;
    const int h    = blockIdx.x;
    const int tid  = threadIdx.x;
    const int lane = tid & 31;
    const int wv   = tid >> 5;
    const int mg   = wv & 3;
    const int nh   = wv >> 2;
    const int o0   = mg * 32;
    const int nc0  = nh * 4;
    const int l15  = lane & 15;
    const int hi   = lane >> 4;
    const int qa   = hi ? 4 : 0;
    const int qb   = hi ? 4 : 0;

    v8f acc0[4], acc1[4];
#pragma unroll
    for (int t = 0; t < 4; ++t) { acc0[t] = {}; acc1[t] = {}; }

    for (int kc = 0; kc < 32; ++kc) {
        __syncthreads();
        for (int idx = tid; idx < 3 * 130 * 16; idx += 256) {
            int q   = idx / 390;
            int p   = idx - q * 390;
            int r   = p / 130;
            int col = p - r * 130;
            int row = h - 1 + r;
            int ww  = col - 1;
            float f0 = 0.f, f1 = 0.f;
            if ((unsigned)row < 64u && (unsigned)ww < 128u) {
                int c = kc * 32 + 2 * q;
                const float* src = (c < 512)
                    ? (in1 + (((long)(b * 512 + c)         * 64 + row) * 128 + ww))
                    : (in2 + (((long)(b * 512 + (c - 512)) * 64 + row) * 128 + ww));
                f0 = src[0];
                f1 = src[64 * 128];
            }
            ldsi[p * 16 + q] = pkh(f0, f1);
        }
        __syncthreads();

#pragma unroll
        for (int tap = 0; tap < 9; ++tap) {
            const int dh = tap / 3, dw = tap % 3;
            Frag A0, A1;
            const unsigned int* wp = wp2 + (long)(kc * 128 + o0 + l15) * 144 + tap * 16;
            A0.q[0] = *(const uint4*)(wp + qa);
            A0.q[1] = *(const uint4*)(wp + qa + 8);
            A1.q[0] = *(const uint4*)(wp + 16 * 144 + qa);
            A1.q[1] = *(const uint4*)(wp + 16 * 144 + qa + 8);
            const int rb = dh * 130;
#pragma unroll
            for (int nt = 0; nt < 4; ++nt) {
                Frag B;
                ldB(B, &ldsi[(rb + (nc0 + nt) * 16 + l15 + dw) * 16 + qb]);
                acc0[nt] = __builtin_amdgcn_wmma_f32_16x16x32_f16(
                    false, A0.v, false, B.v, (short)0, acc0[nt], false, false);
                acc1[nt] = __builtin_amdgcn_wmma_f32_16x16x32_f16(
                    false, A1.v, false, B.v, (short)0, acc1[nt], false, false);
            }
        }
    }

#pragma unroll
    for (int a = 0; a < 2; ++a) {
#pragma unroll
        for (int r = 0; r < 8; ++r) {
            const int o   = o0 + a * 16 + r + hi * 8;
            const float inv = g[o] * rsqrtf(var[o] + 1e-5f);
            const float sh  = (cb[o] - mu[o]) * inv + be[o];
            float* orow = out + ((long)(b * 128 + o) * 64 + h) * 128;
#pragma unroll
            for (int nt = 0; nt < 4; ++nt) {
                float v = (a ? acc1[nt][r] : acc0[nt][r]) * inv + sh;
                orow[(nc0 + nt) * 16 + l15] = v > 0.f ? v : 0.f;
            }
        }
    }
}

// ---------------------------------------------------------------------------
// Sequential row recurrence: for i=1..63: out[:,:,i,:] += conv1d_7(out[:,:,i-1,:]) + b1
// One block per batch, 1024 threads = 32 waves; prev row lives in LDS (f16 pairs).
// ---------------------------------------------------------------------------
__global__ __launch_bounds__(1024) void ffm_rowrec(
    const unsigned int* __restrict__ wp1, const float* __restrict__ b1,
    float* __restrict__ out)
{
    __shared__ alignas(16) unsigned int ldsp[134 * 64];
    const int b    = blockIdx.x;
    const int tid  = threadIdx.x;
    const int lane = tid & 31;
    const int wv   = tid >> 5;
    const int o0   = (wv & 7) * 16;
    const int ns   = (wv >> 3) * 2;
    const int l15  = lane & 15;
    const int hi   = lane >> 4;
    const int qa   = hi ? 4 : 0;
    const int qb   = hi ? 4 : 0;

    for (int idx = tid; idx < 6 * 64; idx += 1024) {
        int cc = idx >> 6, q = idx & 63;
        int col = cc < 3 ? cc : 128 + cc;
        ldsp[col * 64 + q] = 0u;
    }
    const float* base = out + (long)b * 128 * 64 * 128;
    for (int idx = tid; idx < 64 * 128; idx += 1024) {
        int q = idx >> 7, w = idx & 127;
        float f0 = base[(long)(2 * q)     * 8192 + w];
        float f1 = base[(long)(2 * q + 1) * 8192 + w];
        ldsp[(w + 3) * 64 + q] = pkh(f0, f1);
    }
    __syncthreads();

    float bias[8];
#pragma unroll
    for (int r = 0; r < 8; ++r) bias[r] = b1[o0 + r + hi * 8];

    for (int i = 1; i < 64; ++i) {
        v8f acc0 = {}, acc1 = {};
#pragma unroll
        for (int kc = 0; kc < 4; ++kc) {
#pragma unroll
            for (int tap = 0; tap < 7; ++tap) {
                Frag A;
                const unsigned int* wp = wp1 + (long)(kc * 128 + o0 + l15) * 112 + tap * 16;
                A.q[0] = *(const uint4*)(wp + qa);
                A.q[1] = *(const uint4*)(wp + qa + 8);
                Frag B0, B1;
                ldB(B0, &ldsp[(ns * 16 + l15 + tap) * 64 + kc * 16 + qb]);
                ldB(B1, &ldsp[((ns + 1) * 16 + l15 + tap) * 64 + kc * 16 + qb]);
                acc0 = __builtin_amdgcn_wmma_f32_16x16x32_f16(
                    false, A.v, false, B0.v, (short)0, acc0, false, false);
                acc1 = __builtin_amdgcn_wmma_f32_16x16x32_f16(
                    false, A.v, false, B1.v, (short)0, acc1, false, false);
            }
        }
        float vals[8][2];
#pragma unroll
        for (int r = 0; r < 8; ++r) {
            const int o = o0 + r + hi * 8;
            float* yr = out + ((long)(b * 128 + o) * 64 + i) * 128;
#pragma unroll
            for (int t2 = 0; t2 < 2; ++t2) {
                int w = (ns + t2) * 16 + l15;
                float v = yr[w] + (t2 ? acc1[r] : acc0[r]) + bias[r];
                vals[r][t2] = v;
                yr[w] = v;
            }
        }
        __syncthreads();
#pragma unroll
        for (int t2 = 0; t2 < 2; ++t2) {
            int w = (ns + t2) * 16 + l15;
#pragma unroll
            for (int t = 0; t < 4; ++t)
                ldsp[(w + 3) * 64 + (o0 >> 1) + hi * 4 + t] =
                    pkh(vals[2 * t][t2], vals[2 * t + 1][t2]);
        }
        __syncthreads();
    }
}

// ---------------------------------------------------------------------------
extern "C" void kernel_launch(void* const* d_in, const int* in_sizes, int n_in,
                              void* d_out, int out_size, void* d_ws, size_t ws_size,
                              hipStream_t stream) {
    const float* in1 = (const float*)d_in[0];
    const float* in2 = (const float*)d_in[1];
    const float* w2  = (const float*)d_in[2];
    const float* cb  = (const float*)d_in[3];
    const float* g   = (const float*)d_in[4];
    const float* be  = (const float*)d_in[5];
    const float* mu  = (const float*)d_in[6];
    const float* var = (const float*)d_in[7];
    const float* w1  = (const float*)d_in[8];
    const float* b1  = (const float*)d_in[9];
    float* out = (float*)d_out;

    const int N2 = 32 * 128 * 9 * 16;            // packed conv2d weight uints
    const int N1 = 4 * 128 * 7 * 16;             // packed conv1d weight uints
    const long NX = (long)4 * 32 * 64 * 2048;    // packed input uints (64 MB)
    if (ws_size < (size_t)(N2 + N1) * 4) return;
    unsigned int* wp2 = (unsigned int*)d_ws;
    unsigned int* wp1 = wp2 + N2;
    unsigned int* xp  = wp1 + N1;
    const bool use_pk = ws_size >= ((size_t)(N2 + N1) + (size_t)NX) * 4;

    ffm_repack2<<<(N2 + 255) / 256, 256, 0, stream>>>(w2, wp2);
    ffm_repack1<<<(N1 + 255) / 256, 256, 0, stream>>>(w1, wp1);
    if (use_pk) {
        ffm_packin<<<4 * 32 * 64, 256, 0, stream>>>(in1, in2, xp);
        ffm_conv2d_pk<<<dim3(64, 4), 256, 0, stream>>>(xp, wp2, cb, g, be, mu, var, out);
    } else {
        ffm_conv2d<<<dim3(64, 4), 256, 0, stream>>>(in1, in2, wp2, cb, g, be, mu, var, out);
    }
    ffm_rowrec<<<4, 1024, 0, stream>>>(wp1, b1, out);
}
